// Block_14516989461345
// MI455X (gfx1250) — matrix operand order repaired and checked by reading, not compiled, not run on hardware
//
#include <hip/hip_runtime.h>

// ---------------------------------------------------------------------------
// MI455X (gfx1250) transformer block: LN -> QKV -> causal attn -> proj ->
// LN -> MLP.  All GEMMs run through v_wmma_f32_16x16x32_bf16 (wave32).
// ---------------------------------------------------------------------------

typedef unsigned short u16;
typedef __bf16 bf16;
typedef bf16         bf16x16 __attribute__((ext_vector_type(16)));
typedef float        f32x8   __attribute__((ext_vector_type(8)));
typedef unsigned int u32x4   __attribute__((ext_vector_type(4)));

#define DEV __device__ __forceinline__
#define GAS __attribute__((address_space(1)))   // global address space

DEV u16 f32_to_bf16(float f) {
  unsigned int u = __builtin_bit_cast(unsigned int, f);
  unsigned int r = u + 0x7FFFu + ((u >> 16) & 1u);
  return (u16)(r >> 16);
}
DEV float bf16_to_f32(u16 h) {
  unsigned int u = ((unsigned int)h) << 16;
  return __builtin_bit_cast(float, u);
}

// Force global_load/global_store (not flat) in hot paths.
DEV u32x4 ldg4(const u16* p)          { return *(const GAS u32x4*)p; }
DEV float ldgf(const float* p)        { return *(const GAS float*)p; }
DEV void  stg16(u16* p, u16 v)        { *(GAS u16*)p = v; }
DEV void  stgf(float* p, float v)     { *(GAS float*)p = v; }

union Frag { bf16x16 v; u32x4 q[2]; };

DEV f32x8 wmma_bf16(const Frag& a, const Frag& b, f32x8 c) {
  return __builtin_amdgcn_wmma_f32_16x16x32_bf16(
      false, a.v, false, b.v, (short)0, c, false, false);
}

static constexpr int TB = 2048;   // sequence length T
static constexpr int CC = 1024;   // embed dim
static constexpr int HH = 16;     // heads
static constexpr int DD = 64;     // head dim
static constexpr int BB = 4;      // batch
static constexpr int MM = BB * TB;

// ---------------------------------------------------------------------------
// Weight convert + transpose: W (K x N) f32  ->  Wt (N x K) bf16
// ---------------------------------------------------------------------------
__global__ void wt_kernel(const float* __restrict__ W, u16* __restrict__ Wt,
                          int K, int N) {
  size_t o = (size_t)blockIdx.x * blockDim.x + threadIdx.x;
  size_t total = (size_t)K * N;
  if (o >= total) return;
  size_t n = o / K, k = o % K;
  Wt[o] = f32_to_bf16(W[k * (size_t)N + n]);
}

// ---------------------------------------------------------------------------
// LayerNorm over 1024-wide rows, output bf16.  One block (256 thr) per row.
// ---------------------------------------------------------------------------
__global__ void ln_kernel(const float* __restrict__ x,
                          const float* __restrict__ g,
                          const float* __restrict__ be,
                          u16* __restrict__ out) {
  __shared__ float s1[256], s2[256], smu, srv;
  size_t row = blockIdx.x;
  const float* xr = x + row * CC;
  int t = threadIdx.x;
  float4 v = *(const float4*)(xr + t * 4);
  float s = v.x + v.y + v.z + v.w;
  float ss = v.x * v.x + v.y * v.y + v.z * v.z + v.w * v.w;
  s1[t] = s; s2[t] = ss;
  __syncthreads();
  for (int off = 128; off > 0; off >>= 1) {
    if (t < off) { s1[t] += s1[t + off]; s2[t] += s2[t + off]; }
    __syncthreads();
  }
  if (t == 0) {
    float mu = s1[0] * (1.0f / CC);
    float var = s2[0] * (1.0f / CC) - mu * mu;
    smu = mu;
    srv = __frsqrt_rn(var + 1e-5f);
  }
  __syncthreads();
  float mu = smu, rv = srv;
  u16* orow = out + row * CC;
  int c = t * 4;
  orow[c + 0] = f32_to_bf16((v.x - mu) * rv * g[c + 0] + be[c + 0]);
  orow[c + 1] = f32_to_bf16((v.y - mu) * rv * g[c + 1] + be[c + 1]);
  orow[c + 2] = f32_to_bf16((v.z - mu) * rv * g[c + 2] + be[c + 2]);
  orow[c + 3] = f32_to_bf16((v.w - mu) * rv * g[c + 3] + be[c + 3]);
}

// ---------------------------------------------------------------------------
// Generic bf16 WMMA GEMM.  A: (M x K) bf16 row-major.  Bt: (N x K) bf16.
// One wave computes a 64x64 tile (4x4 WMMA accumulators), 4 waves / block.
// B-fragments streamed one at a time: 16 acc + 4 A + 1 B + addrs < 256 VGPRs.
// out_mode: 0 = bf16 row-major (+bias,+relu)
//           1 = bf16 head-split (B,H,T,D)
//           2 = bf16 head-split transposed (B,H,D,T)    (for V)
//           3 = f32  row-major, += bias[n] + residual   (residual f32 MxN)
// ---------------------------------------------------------------------------
__global__ __launch_bounds__(128, 1)
void gemm_wmma(const u16* __restrict__ A, const u16* __restrict__ Bt,
               int M, int N, int K,
               const float* __restrict__ bias,
               const float* __restrict__ residual,
               float* __restrict__ outF, u16* __restrict__ outH,
               int out_mode, int do_relu, int T) {
  const int lane = threadIdx.x & 31;
  const int wave = threadIdx.x >> 5;
  const int half = lane >> 4;
  const int lr   = lane & 15;
  const int m0 = (blockIdx.y * 4 + wave) * 64;
  const int n0 = blockIdx.x * 64;
  if (m0 >= M) return;

  f32x8 acc[4][4];
#pragma unroll
  for (int i = 0; i < 4; ++i)
#pragma unroll
    for (int j = 0; j < 4; ++j) acc[i][j] = (f32x8)0.0f;

  // Per-wave row/col base pointers (advance by 32 elements per K-step).
  const u16* aptr[4];
#pragma unroll
  for (int mi = 0; mi < 4; ++mi)
    aptr[mi] = A + (size_t)(m0 + mi * 16 + lr) * K;
  const u16* bptr[4];
#pragma unroll
  for (int ni = 0; ni < 4; ++ni)
    bptr[ni] = Bt + (size_t)(n0 + ni * 16 + lr) * K;

#pragma unroll 1
  for (int k0 = 0; k0 < K; k0 += 32) {
    Frag a[4];
#pragma unroll
    for (int mi = 0; mi < 4; ++mi) {
      a[mi].q[0] = ldg4(aptr[mi] + half * 8);
      a[mi].q[1] = ldg4(aptr[mi] + 16 + half * 8);
      aptr[mi] += 32;
    }
#pragma unroll
    for (int ni = 0; ni < 4; ++ni) {
      Frag b;
      b.q[0] = ldg4(bptr[ni] + half * 8);
      b.q[1] = ldg4(bptr[ni] + 16 + half * 8);
      bptr[ni] += 32;
#pragma unroll
      for (int mi = 0; mi < 4; ++mi)
        acc[mi][ni] = wmma_bf16(a[mi], b, acc[mi][ni]);
    }
  }

  // Epilogue: C/D layout: VGPR j -> row (j + 8*half), col = lane%16.
#pragma unroll
  for (int mi = 0; mi < 4; ++mi) {
#pragma unroll
    for (int j = 0; j < 8; ++j) {
      int m = m0 + mi * 16 + j + half * 8;
      int bidx = m / T, tt = m % T;
#pragma unroll
      for (int ni = 0; ni < 4; ++ni) {
        int n = n0 + ni * 16 + lr;
        float val = acc[mi][ni][j];
        if (out_mode == 0) {
          if (bias) val += ldgf(bias + n);
          if (do_relu) val = fmaxf(val, 0.0f);
          stg16(outH + (size_t)m * N + n, f32_to_bf16(val));
        } else if (out_mode == 1) {
          int h = n >> 6, d = n & 63;
          stg16(outH + (((size_t)bidx * HH + h) * T + tt) * DD + d,
                f32_to_bf16(val));
        } else if (out_mode == 2) {
          int h = n >> 6, d = n & 63;
          stg16(outH + (((size_t)bidx * HH + h) * DD + d) * T + tt,
                f32_to_bf16(val));
        } else {
          val += ldgf(bias + n) + ldgf(residual + (size_t)m * N + n);
          stgf(outF + (size_t)m * N + n, val);
        }
      }
    }
  }
}

// ---------------------------------------------------------------------------
// Flash-style causal attention.  One wave handles 16 q-rows of one (b,h).
// q,k: (B,H,T,D) bf16.  vT: (B,H,D,T) bf16.  out: (B,T,C) bf16.
// ---------------------------------------------------------------------------
__global__ __launch_bounds__(128, 1)
void attn_kernel(const u16* __restrict__ qb,
                 const u16* __restrict__ kb,
                 const u16* __restrict__ vtb,
                 u16* __restrict__ outh) {
  constexpr int LSTR = 40;                     // LDS row stride (bf16 elems)
  __shared__ u16 Plds[4][16 * LSTR];

  const int lane = threadIdx.x & 31;
  const int wave = threadIdx.x >> 5;
  const int half = lane >> 4;
  const int lr   = lane & 15;
  const int bh = blockIdx.y;
  const int b  = bh >> 4;
  const int h  = bh & 15;
  const int t0 = (blockIdx.x * 4 + wave) * 16;

  const u16* qh = qb  + ((size_t)bh * TB) * DD;
  const u16* kh = kb  + ((size_t)bh * TB) * DD;
  const u16* vh = vtb + ((size_t)bh * DD) * TB;

  // q A-fragments (16 rows x 64 d -> two K=32 fragments)
  Frag aqlo, aqhi;
  {
    const u16* qp = qh + (size_t)(t0 + lr) * DD;
    aqlo.q[0] = ldg4(qp + half * 8);
    aqlo.q[1] = ldg4(qp + 16 + half * 8);
    aqhi.q[0] = ldg4(qp + 32 + half * 8);
    aqhi.q[1] = ldg4(qp + 48 + half * 8);
  }

  f32x8 o[4];
#pragma unroll
  for (int i = 0; i < 4; ++i) o[i] = (f32x8)0.0f;
  float mrow[8], lrow[8];
#pragma unroll
  for (int j = 0; j < 8; ++j) { mrow[j] = -1e30f; lrow[j] = 0.0f; }

#pragma unroll 1
  for (int s0 = 0; s0 < t0 + 16; s0 += 32) {
    // ---- scores S = q . k^T  (16 x 32) ----
    f32x8 S[2];
#pragma unroll
    for (int nb = 0; nb < 2; ++nb) {
      Frag bk0, bk1;
      const u16* kp = kh + (size_t)(s0 + nb * 16 + lr) * DD;
      bk0.q[0] = ldg4(kp + half * 8);
      bk0.q[1] = ldg4(kp + 16 + half * 8);
      bk1.q[0] = ldg4(kp + 32 + half * 8);
      bk1.q[1] = ldg4(kp + 48 + half * 8);
      S[nb] = wmma_bf16(aqlo, bk0, (f32x8)0.0f);
      S[nb] = wmma_bf16(aqhi, bk1, S[nb]);
    }

    // ---- online softmax (row stats live per (j, lane-half)) ----
#pragma unroll
    for (int j = 0; j < 8; ++j) {
      int tg = t0 + j + half * 8;
      float s0v = S[0][j] * 0.125f;          // 1/sqrt(64)
      float s1v = S[1][j] * 0.125f;
      if (s0 + lr      > tg) s0v = -1e30f;
      if (s0 + 16 + lr > tg) s1v = -1e30f;
      float rmax = fmaxf(s0v, s1v);
#pragma unroll
      for (int off = 1; off < 16; off <<= 1)
        rmax = fmaxf(rmax, __shfl_xor(rmax, off, 32));
      float mnew = fmaxf(mrow[j], rmax);
      float corr = __expf(mrow[j] - mnew);
      float p0 = __expf(s0v - mnew);
      float p1 = __expf(s1v - mnew);
      float rsum = p0 + p1;
#pragma unroll
      for (int off = 1; off < 16; off <<= 1)
        rsum += __shfl_xor(rsum, off, 32);
      lrow[j] = lrow[j] * corr + rsum;
      mrow[j] = mnew;
#pragma unroll
      for (int nb = 0; nb < 4; ++nb) o[nb][j] *= corr;
      int prow = j + half * 8;
      Plds[wave][prow * LSTR + lr]      = f32_to_bf16(p0);
      Plds[wave][prow * LSTR + 16 + lr] = f32_to_bf16(p1);
    }

    // DS ops are in-order within a wave; make the dependency explicit.
    asm volatile("s_wait_dscnt 0" ::: "memory");

    // ---- P as A-fragment (16 x 32) from LDS ----
    Frag pf;
    pf.q[0] = *(const u32x4*)(&Plds[wave][lr * LSTR + half * 8]);
    pf.q[1] = *(const u32x4*)(&Plds[wave][lr * LSTR + 16 + half * 8]);

    // ---- O += P . V   (V^T gives contiguous-s B-fragments) ----
#pragma unroll
    for (int nb = 0; nb < 4; ++nb) {
      Frag bv;
      const u16* vp = vh + (size_t)(nb * 16 + lr) * TB + s0 + half * 8;
      bv.q[0] = ldg4(vp);
      bv.q[1] = ldg4(vp + 16);
      o[nb] = wmma_bf16(pf, bv, o[nb]);
    }
  }

  // ---- normalize + write (B,T,C) bf16 ----
#pragma unroll
  for (int j = 0; j < 8; ++j) {
    int t = t0 + j + half * 8;
    float inv = 1.0f / lrow[j];
#pragma unroll
    for (int nb = 0; nb < 4; ++nb) {
      int d = nb * 16 + lr;
      stg16(outh + ((size_t)b * TB + t) * CC + h * DD + d,
            f32_to_bf16(o[nb][j] * inv));
    }
  }
}

// ---------------------------------------------------------------------------
extern "C" void kernel_launch(void* const* d_in, const int* in_sizes, int n_in,
                              void* d_out, int out_size, void* d_ws,
                              size_t ws_size, hipStream_t stream) {
  (void)in_sizes; (void)n_in; (void)out_size; (void)ws_size;
  const float* x     = (const float*)d_in[0];
  const float* Wk    = (const float*)d_in[1];
  const float* Wq    = (const float*)d_in[2];
  const float* Wv    = (const float*)d_in[3];
  const float* Wproj = (const float*)d_in[4];
  const float* bproj = (const float*)d_in[5];
  const float* W1    = (const float*)d_in[6];
  const float* b1    = (const float*)d_in[7];
  const float* W2    = (const float*)d_in[8];
  const float* b2    = (const float*)d_in[9];
  const float* g1    = (const float*)d_in[10];
  const float* be1   = (const float*)d_in[11];
  const float* g2    = (const float*)d_in[12];
  const float* be2   = (const float*)d_in[13];

  char* ws = (char*)d_ws;
  const size_t MB = 1ull << 20;
  u16*   Wqt  = (u16*)(ws + 0 * MB);     //  2 MB
  u16*   Wkt  = (u16*)(ws + 2 * MB);     //  2 MB
  u16*   Wvt  = (u16*)(ws + 4 * MB);     //  2 MB
  u16*   Wpt  = (u16*)(ws + 6 * MB);     //  2 MB
  u16*   W1t  = (u16*)(ws + 8 * MB);     //  8 MB
  u16*   W2t  = (u16*)(ws + 16 * MB);    //  8 MB
  float* xres = (float*)(ws + 24 * MB);  // 32 MB
  u16*   regA = (u16*)(ws + 56 * MB);    // 16 MB: attn_out, later h2
  u16*   hbuf = (u16*)(ws + 72 * MB);    // 16 MB (LN1 out; later part of ff1)
  u16*   qb   = (u16*)(ws + 88 * MB);    // 16 MB
  u16*   kbuf = (u16*)(ws + 104 * MB);   // 16 MB
  u16*   vtb  = (u16*)(ws + 120 * MB);   // 16 MB   (total 136 MB)
  u16*   ff1  = hbuf;                    // 64 MB region overlaying h,q,k,v

  // --- weight transposes to bf16 (N x K) ---
  wt_kernel<<<(1024 * 1024) / 256, 256, 0, stream>>>(Wq, Wqt, 1024, 1024);
  wt_kernel<<<(1024 * 1024) / 256, 256, 0, stream>>>(Wk, Wkt, 1024, 1024);
  wt_kernel<<<(1024 * 1024) / 256, 256, 0, stream>>>(Wv, Wvt, 1024, 1024);
  wt_kernel<<<(1024 * 1024) / 256, 256, 0, stream>>>(Wproj, Wpt, 1024, 1024);
  wt_kernel<<<(1024 * 4096) / 256, 256, 0, stream>>>(W1, W1t, 1024, 4096);
  wt_kernel<<<(4096 * 1024) / 256, 256, 0, stream>>>(W2, W2t, 4096, 1024);

  // --- LN1 ---
  ln_kernel<<<MM, 256, 0, stream>>>(x, g1, be1, hbuf);

  // --- QKV projections ---
  dim3 blk(128);
  gemm_wmma<<<dim3(1024 / 64, MM / 256), blk, 0, stream>>>(
      hbuf, Wqt, MM, 1024, 1024, nullptr, nullptr, nullptr, qb, 1, 0, TB);
  gemm_wmma<<<dim3(1024 / 64, MM / 256), blk, 0, stream>>>(
      hbuf, Wkt, MM, 1024, 1024, nullptr, nullptr, nullptr, kbuf, 1, 0, TB);
  gemm_wmma<<<dim3(1024 / 64, MM / 256), blk, 0, stream>>>(
      hbuf, Wvt, MM, 1024, 1024, nullptr, nullptr, nullptr, vtb, 2, 0, TB);

  // --- causal attention ---
  attn_kernel<<<dim3(TB / 64, BB * HH), blk, 0, stream>>>(qb, kbuf, vtb, regA);

  // --- output projection + residual ---
  gemm_wmma<<<dim3(1024 / 64, MM / 256), blk, 0, stream>>>(
      regA, Wpt, MM, 1024, 1024, bproj, x, xres, nullptr, 3, 0, TB);

  // --- LN2 (h2 overwrites attn_out region) ---
  ln_kernel<<<MM, 256, 0, stream>>>(xres, g2, be2, regA);

  // --- MLP ---
  gemm_wmma<<<dim3(4096 / 64, MM / 256), blk, 0, stream>>>(
      regA, W1t, MM, 4096, 1024, b1, nullptr, nullptr, ff1, 0, 1, TB);
  gemm_wmma<<<dim3(1024 / 64, MM / 256), blk, 0, stream>>>(
      ff1, W2t, MM, 1024, 4096, b2, xres, (float*)d_out, nullptr, 3, 0, TB);
}
